// CrossPersonInteraction_54666343743509
// MI455X (gfx1250) — matrix operand order repaired and checked by reading, not compiled, not run on hardware
//
#include <hip/hip_runtime.h>
#include <math.h>

typedef __attribute__((ext_vector_type(16))) __bf16 v16bf;
typedef __attribute__((ext_vector_type(8)))  __bf16 v8bf;
typedef __attribute__((ext_vector_type(8)))  float  v8f;
typedef __attribute__((ext_vector_type(4)))  float  v4f;

#define DD      256
#define KJ      25
#define TOKENS  128000      // 8*128*5*25
#define MTILES  8000        // TOKENS/16

__constant__ int c_joints[5][7] = {
  {0,1,20,2,3,0,0},
  {20,4,5,6,7,21,22},
  {20,8,9,10,11,23,24},
  {0,12,13,14,15,0,0},
  {0,16,17,18,19,0,0}};
__constant__ int c_map25[3] = {0,3,4};   // parts with l=5 (L=25)
__constant__ int c_map35[2] = {1,2};     // parts with l=7 (L=35)

// ---------------- Kernel 0: zero accumulator, convert weights to bf16 -------
__global__ void __launch_bounds__(256)
init_kernel(const float* __restrict__ wq, const float* __restrict__ wk,
            const float* __restrict__ wv, const float* __restrict__ wo,
            __bf16* __restrict__ wbf, float* __restrict__ accum) {
  size_t i = (size_t)blockIdx.x * blockDim.x + threadIdx.x;
  if (i < (size_t)TOKENS * DD) accum[i] = 0.0f;
  if (i < 4u * 65536u) {
    int wsel = (int)(i >> 16);
    int off  = (int)(i & 65535);
    const float* src = (wsel == 0) ? wq : (wsel == 1) ? wk : (wsel == 2) ? wv : wo;
    wbf[i] = (__bf16)src[off];
  }
}

// ---------------- Kernel 1: Q,K,V = X * W^T + b  (bf16 WMMA, f32 accum) -----
__global__ void __launch_bounds__(256)
qkv_gemm(const float* __restrict__ x, const __bf16* __restrict__ wbf,
         const float* __restrict__ bq, const float* __restrict__ bk,
         const float* __restrict__ bv,
         __bf16* __restrict__ qb, __bf16* __restrict__ kbuf, __bf16* __restrict__ vb) {
  const int lane = threadIdx.x & 31;
  const int wave = threadIdx.x >> 5;
  const int half = lane >> 4;
  const int mr   = lane & 15;
  const int tileM = blockIdx.x;
  const int r = tileM * 16 + mr;           // A row (token) this lane loads

  const __bf16* wqb = wbf;
  const __bf16* wkb = wbf + 65536;
  const __bf16* wvb = wbf + 2 * 65536;

  const int nt0 = wave * 2, nt1 = wave * 2 + 1;
  const int nrow0 = nt0 * 16 + mr;
  const int nrow1 = nt1 * 16 + mr;

  v8f aq0 = {}, aq1 = {}, ak0 = {}, ak1 = {}, av0 = {}, av1 = {};

  for (int ks = 0; ks < 8; ++ks) {
    const int kb0 = ks * 32;
    // A tile 16x32 (f32 -> bf16 on the fly), contiguous k order (k = 16*half + s)
    const float* xp = x + (size_t)r * DD + kb0 + half * 16;
    v4f f0 = *(const v4f*)(xp);
    v4f f1 = *(const v4f*)(xp + 4);
    v4f f2 = *(const v4f*)(xp + 8);
    v4f f3 = *(const v4f*)(xp + 12);
    v16bf a;
#pragma unroll
    for (int e = 0; e < 4; ++e) {
      a[e]      = (__bf16)f0[e];
      a[4 + e]  = (__bf16)f1[e];
      a[8 + e]  = (__bf16)f2[e];
      a[12 + e] = (__bf16)f3[e];
    }
    const int bofs = kb0 + half * 16;
    v16bf bQ0 = *(const v16bf*)(wqb + (size_t)nrow0 * DD + bofs);
    v16bf bQ1 = *(const v16bf*)(wqb + (size_t)nrow1 * DD + bofs);
    v16bf bK0 = *(const v16bf*)(wkb + (size_t)nrow0 * DD + bofs);
    v16bf bK1 = *(const v16bf*)(wkb + (size_t)nrow1 * DD + bofs);
    v16bf bV0 = *(const v16bf*)(wvb + (size_t)nrow0 * DD + bofs);
    v16bf bV1 = *(const v16bf*)(wvb + (size_t)nrow1 * DD + bofs);

    aq0 = __builtin_amdgcn_wmma_f32_16x16x32_bf16(false, a, false, bQ0, (short)0, aq0, false, false);
    aq1 = __builtin_amdgcn_wmma_f32_16x16x32_bf16(false, a, false, bQ1, (short)0, aq1, false, false);
    ak0 = __builtin_amdgcn_wmma_f32_16x16x32_bf16(false, a, false, bK0, (short)0, ak0, false, false);
    ak1 = __builtin_amdgcn_wmma_f32_16x16x32_bf16(false, a, false, bK1, (short)0, ak1, false, false);
    av0 = __builtin_amdgcn_wmma_f32_16x16x32_bf16(false, a, false, bV0, (short)0, av0, false, false);
    av1 = __builtin_amdgcn_wmma_f32_16x16x32_bf16(false, a, false, bV1, (short)0, av1, false, false);
  }

  const int col0 = nt0 * 16 + mr;
  const int col1 = nt1 * 16 + mr;
  const float bq0 = bq[col0], bq1 = bq[col1];
  const float bk0s = bk[col0], bk1s = bk[col1];
  const float bv0s = bv[col0], bv1s = bv[col1];
#pragma unroll
  for (int i = 0; i < 8; ++i) {
    const int rr = tileM * 16 + i + half * 8;  // D row: M = i + half*8
    const size_t o0 = (size_t)rr * DD + col0;
    const size_t o1 = (size_t)rr * DD + col1;
    qb[o0]   = (__bf16)(aq0[i] + bq0);
    qb[o1]   = (__bf16)(aq1[i] + bq1);
    kbuf[o0] = (__bf16)(ak0[i] + bk0s);
    kbuf[o1] = (__bf16)(ak1[i] + bk1s);
    vb[o0]   = (__bf16)(av0[i] + bv0s);
    vb[o1]   = (__bf16)(av1[i] + bv1s);
  }
}

// ---------------- Kernel 2: per (batch, part) attention, scatter-accumulate -
// Templated on joints-per-part LJ (5 -> L=25, 7 -> L=35) so all loops unroll,
// K B-tiles live in registers, and the per-head V slice is staged transposed
// in LDS so every P*V B-tile is one aligned 32-byte ds load.
template<int LJ>
__global__ void __launch_bounds__(128)
attn_kernel(const __bf16* __restrict__ qb, const __bf16* __restrict__ kbuf,
            const __bf16* __restrict__ vb, float* __restrict__ accum) {
  constexpr int L   = 5 * LJ;               // 25 or 35
  constexpr int NQT = (L + 15) / 16;        // 2 or 3
  constexpr int KPV = (NQT == 2) ? 32 : 64; // zero-padded K extent for P*V
  constexpr int NKS = KPV / 32;

  __shared__ alignas(32) __bf16 scb[4][48 * 48]; // bf16 scores, per wave
  __shared__ alignas(32) __bf16 pmm[4][48 * 64]; // softmax probs, per wave
  __shared__ alignas(32) __bf16 vst[4][32 * 64]; // V slice, [ch][s] per wave
  __shared__ int seq[64];

  const int b = blockIdx.x;                 // 0..1023 (n*t)
  const int part = (LJ == 5) ? c_map25[blockIdx.y] : c_map35[blockIdx.y];
  const int tid = threadIdx.x;

  if (tid < 64) {
    int v = -1;
    if (tid < L) {
      int mm = tid / LJ;
      int jj = c_joints[part][tid % LJ];
      v = b * 125 + mm * 25 + jj;           // token index into [.,125,256]
    }
    seq[tid] = v;
  }
  __syncthreads();

  const int lane = tid & 31;
  const int wave = tid >> 5;
  const int half = lane >> 4;
  const int mr   = lane & 15;

  __bf16* scw  = &scb[wave][0];
  __bf16* pw   = &pmm[wave][0];
  __bf16* vstw = &vst[wave][0];
  const float scale = 0.17677669529663687f;  // 1/sqrt(32)

  for (int hi = 0; hi < 2; ++hi) {
    const int h  = wave + hi * 4;            // head
    const int c0 = h * 32;

    // ---- stage V slice transposed into LDS: vstw[ch*64 + s] ----
    {
      const int ch = lane;
#pragma unroll
      for (int s = 0; s < 64; ++s) {
        __bf16 v = (__bf16)0.0f;
        if (s < L) v = vb[(size_t)seq[s] * DD + c0 + ch];  // coalesced 64B/wave
        vstw[ch * 64 + s] = v;
      }
    }

    // ---- K B-tiles cached in registers for the whole head ----
    v16bf bKt[NQT];
#pragma unroll
    for (int kt = 0; kt < NQT; ++kt) {
      const int s2 = kt * 16 + mr;
      const int tok2 = seq[s2];
      if (tok2 >= 0) {
        bKt[kt] = *(const v16bf*)(kbuf + (size_t)tok2 * DD + c0 + half * 16);
      } else {
#pragma unroll
        for (int e = 0; e < 16; ++e) bKt[kt][e] = (__bf16)0.0f;
      }
    }

    // ---- scores = Q K^T * scale (dh=32 -> one WMMA per 16x16 tile) ----
#pragma unroll
    for (int qt = 0; qt < NQT; ++qt) {
      const int s = qt * 16 + mr;
      const int tok = seq[s];
      v16bf aQ;
      if (tok >= 0) {
        const __bf16* qp = qb + (size_t)tok * DD + c0 + half * 16;
        v8bf q0 = *(const v8bf*)qp;
        v8bf q1 = *(const v8bf*)(qp + 8);
#pragma unroll
        for (int e = 0; e < 8; ++e) { aQ[e] = q0[e]; aQ[8 + e] = q1[e]; }
      } else {
#pragma unroll
        for (int e = 0; e < 16; ++e) aQ[e] = (__bf16)0.0f;
      }
#pragma unroll
      for (int kt = 0; kt < NQT; ++kt) {
        v8f z = {};
        v8f sres = __builtin_amdgcn_wmma_f32_16x16x32_bf16(false, aQ, false, bKt[kt], (short)0, z, false, false);
#pragma unroll
        for (int i = 0; i < 8; ++i) {
          const int row = qt * 16 + i + half * 8;
          scw[row * 48 + kt * 16 + mr] = (__bf16)(sres[i] * scale);
        }
      }
    }

    // ---- row softmax -> P (bf16, stride 64, zero padded) ----
#pragma unroll
    for (int rr2 = 0; rr2 < 2; ++rr2) {
      const int r = lane + rr2 * 32;
      if (r < NQT * 16) {
        if (r < L) {
          float m = -1e30f;
#pragma unroll
          for (int c = 0; c < L; ++c) m = fmaxf(m, (float)scw[r * 48 + c]);
          float sum = 0.0f;
#pragma unroll
          for (int c = 0; c < L; ++c) sum += __expf((float)scw[r * 48 + c] - m);
          const float inv = 1.0f / sum;
#pragma unroll
          for (int c = 0; c < L; ++c)
            pw[r * 64 + c] = (__bf16)(__expf((float)scw[r * 48 + c] - m) * inv);
#pragma unroll
          for (int c = L; c < KPV; ++c) pw[r * 64 + c] = (__bf16)0.0f;
        } else {
#pragma unroll
          for (int c = 0; c < KPV; ++c) pw[r * 64 + c] = (__bf16)0.0f;
        }
      }
    }

    // ---- O = P * V, scatter-add into accumulator ----
#pragma unroll
    for (int qt = 0; qt < NQT; ++qt) {
      // A tiles of P for this query tile, hoisted across the two N tiles
      v16bf aPt[NKS];
#pragma unroll
      for (int ks = 0; ks < NKS; ++ks) {
        const int srow = qt * 16 + mr;
        const __bf16* pp = pw + srow * 64 + ks * 32 + half * 16;
        v8bf p0 = *(const v8bf*)pp;
        v8bf p1 = *(const v8bf*)(pp + 8);
#pragma unroll
        for (int e = 0; e < 8; ++e) { aPt[ks][e] = p0[e]; aPt[ks][8 + e] = p1[e]; }
      }
#pragma unroll
      for (int nt = 0; nt < 2; ++nt) {
        v8f acc = {};
#pragma unroll
        for (int ks = 0; ks < NKS; ++ks) {
          // B tile: one aligned 32B LDS load per lane (V stored [ch][s])
          v16bf bV = *(const v16bf*)(vstw + (size_t)(nt * 16 + mr) * 64 + ks * 32 + half * 16);
          acc = __builtin_amdgcn_wmma_f32_16x16x32_bf16(false, aPt[ks], false, bV, (short)0, acc, false, false);
        }
#pragma unroll
        for (int i = 0; i < 8; ++i) {
          const int s = qt * 16 + i + half * 8;
          const int tok = seq[s];
          if (tok >= 0)
            atomicAdd(&accum[(size_t)tok * DD + c0 + nt * 16 + mr], acc[i]);
        }
      }
    }
  }
}

// ---------------- Kernel 3: out = (accum/count) * Wo^T + bo -----------------
__global__ void __launch_bounds__(256)
out_gemm(const float* __restrict__ accum, const __bf16* __restrict__ wobf,
         const float* __restrict__ bo, float* __restrict__ out) {
  const int lane = threadIdx.x & 31;
  const int wave = threadIdx.x >> 5;
  const int half = lane >> 4;
  const int mr   = lane & 15;
  const int tileM = blockIdx.x;
  const int r = tileM * 16 + mr;

  const int jj = r % KJ;
  const float rs = (jj == 0 || jj == 20) ? (1.0f / 3.0f) : 1.0f;  // 1/count

  const int nt0 = wave * 2, nt1 = wave * 2 + 1;
  const int nrow0 = nt0 * 16 + mr;
  const int nrow1 = nt1 * 16 + mr;

  v8f a0 = {}, a1 = {};
  for (int ks = 0; ks < 8; ++ks) {
    const int kb0 = ks * 32;
    const float* xp = accum + (size_t)r * DD + kb0 + half * 16;
    v4f f0 = *(const v4f*)(xp);
    v4f f1 = *(const v4f*)(xp + 4);
    v4f f2 = *(const v4f*)(xp + 8);
    v4f f3 = *(const v4f*)(xp + 12);
    v16bf a;
#pragma unroll
    for (int e = 0; e < 4; ++e) {
      a[e]      = (__bf16)(f0[e] * rs);
      a[4 + e]  = (__bf16)(f1[e] * rs);
      a[8 + e]  = (__bf16)(f2[e] * rs);
      a[12 + e] = (__bf16)(f3[e] * rs);
    }
    const int bofs = kb0 + half * 16;
    v16bf b0 = *(const v16bf*)(wobf + (size_t)nrow0 * DD + bofs);
    v16bf b1 = *(const v16bf*)(wobf + (size_t)nrow1 * DD + bofs);
    a0 = __builtin_amdgcn_wmma_f32_16x16x32_bf16(false, a, false, b0, (short)0, a0, false, false);
    a1 = __builtin_amdgcn_wmma_f32_16x16x32_bf16(false, a, false, b1, (short)0, a1, false, false);
  }

  const int col0 = nt0 * 16 + mr;
  const int col1 = nt1 * 16 + mr;
  const float bo0 = bo[col0], bo1 = bo[col1];
#pragma unroll
  for (int i = 0; i < 8; ++i) {
    const int rr = tileM * 16 + i + half * 8;
    out[(size_t)rr * DD + col0] = a0[i] + bo0;
    out[(size_t)rr * DD + col1] = a1[i] + bo1;
  }
}

// ---------------- Host launcher --------------------------------------------
extern "C" void kernel_launch(void* const* d_in, const int* in_sizes, int n_in,
                              void* d_out, int out_size, void* d_ws, size_t ws_size,
                              hipStream_t stream) {
  const float* x  = (const float*)d_in[0];
  const float* wq = (const float*)d_in[1];
  const float* bq = (const float*)d_in[2];
  const float* wk = (const float*)d_in[3];
  const float* bk = (const float*)d_in[4];
  const float* wv = (const float*)d_in[5];
  const float* bv = (const float*)d_in[6];
  const float* wo = (const float*)d_in[7];
  const float* bo = (const float*)d_in[8];

  char* ws = (char*)d_ws;
  const size_t QKV_BYTES = (size_t)TOKENS * DD * 2;      // 65,536,000 each
  __bf16* wbf  = (__bf16*)ws;                            // 524,288 B
  __bf16* qb   = (__bf16*)(ws + 524288);
  __bf16* kb   = (__bf16*)(ws + 524288 + QKV_BYTES);
  __bf16* vb   = (__bf16*)(ws + 524288 + 2 * QKV_BYTES);
  float*  accum = (float*)(ws + 524288 + 3 * QKV_BYTES); // 131,072,000 B
  float*  out  = (float*)d_out;

  init_kernel<<<(TOKENS * DD) / 256, 256, 0, stream>>>(wq, wk, wv, wo, wbf, accum);
  qkv_gemm<<<MTILES, 256, 0, stream>>>(x, wbf, bq, bk, bv, qb, kb, vb);
  attn_kernel<5><<<dim3(1024, 3), 128, 0, stream>>>(qb, kb, vb, accum);
  attn_kernel<7><<<dim3(1024, 2), 128, 0, stream>>>(qb, kb, vb, accum);
  out_gemm<<<MTILES, 256, 0, stream>>>(accum, wbf + 3 * 65536, bo, out);
}
